// GraphConvolutionBlock_6227702579618
// MI455X (gfx1250) — matrix operand order repaired and checked by reading, not compiled, not run on hardware
//
#include <hip/hip_runtime.h>
#include <hip/hip_bf16.h>

#define HID      64
#define IN_DIM   92
#define EDGE_DIM 41
#define KDIM     192      // 3*HID concat width
#define ZS       200      // padded z-tile row stride (bf16 elems): 400B = 100 dwords, bank-conflict free
#define KPAD     200      // padded transposed-weight row stride (bf16 elems)
#define NEG_SLOPE 0.1f
#define BN_EPS    1e-5f

typedef __bf16 v8bf  __attribute__((ext_vector_type(8)));
typedef __bf16 v16bf __attribute__((ext_vector_type(16)));
typedef float  v8f   __attribute__((ext_vector_type(8)));

// pack 4 fp32 -> 4 bf16 in a uint2 (for 8B LDS stores)
__device__ __forceinline__ uint2 bf4(float4 v) {
    union { __bf16 b[4]; uint2 u; } r;
    r.b[0] = (__bf16)v.x; r.b[1] = (__bf16)v.y;
    r.b[2] = (__bf16)v.z; r.b[3] = (__bf16)v.w;
    return r.u;
}

__device__ __forceinline__ v16bf cat16(v8bf lo, v8bf hi) {
    return __builtin_shufflevector(lo, hi, 0,1,2,3,4,5,6,7,8,9,10,11,12,13,14,15);
}

// ---------------- node embed: h = leaky_relu(bn(x @ W_in + b)) ----------------
__global__ void k_node_embed(const float* __restrict__ x, const float* __restrict__ W,
                             const float* __restrict__ b, const float* __restrict__ g,
                             const float* __restrict__ be, const float* __restrict__ rm,
                             const float* __restrict__ rv, float* __restrict__ h, int n) {
    __shared__ float xs[4][IN_DIM];
    const int nb = blockIdx.x * 4;
    for (int i = threadIdx.x; i < 4 * IN_DIM; i += blockDim.x) {
        int r = i / IN_DIM, c = i % IN_DIM;
        if (nb + r < n) xs[r][c] = x[(size_t)(nb + r) * IN_DIM + c];
    }
    __syncthreads();
    const int r = threadIdx.x >> 6, col = threadIdx.x & 63;
    if (nb + r >= n) return;
    float acc = b[col];
    #pragma unroll 4
    for (int k = 0; k < IN_DIM; ++k) acc = fmaf(xs[r][k], W[k * HID + col], acc);
    float y = (acc - rm[col]) * rsqrtf(rv[col] + BN_EPS) * g[col] + be[col];
    h[(size_t)(nb + r) * HID + col] = (y >= 0.f) ? y : NEG_SLOPE * y;
}

// ------- edge embed: e = leaky_relu(bn(ea @ W_e + b)), stored bf16 for WMMA -------
__global__ void k_edge_embed(const float* __restrict__ ea, const float* __restrict__ W,
                             const float* __restrict__ b, const float* __restrict__ g,
                             const float* __restrict__ be, const float* __restrict__ rm,
                             const float* __restrict__ rv, __bf16* __restrict__ ebf, int n) {
    __shared__ float es[4][EDGE_DIM];
    const int nb = blockIdx.x * 4;
    for (int i = threadIdx.x; i < 4 * EDGE_DIM; i += blockDim.x) {
        int r = i / EDGE_DIM, c = i % EDGE_DIM;
        if (nb + r < n) es[r][c] = ea[(size_t)(nb + r) * EDGE_DIM + c];
    }
    __syncthreads();
    const int r = threadIdx.x >> 6, col = threadIdx.x & 63;
    if (nb + r >= n) return;
    float acc = b[col];
    #pragma unroll
    for (int k = 0; k < EDGE_DIM; ++k) acc = fmaf(es[r][k], W[k * HID + col], acc);
    float y = (acc - rm[col]) * rsqrtf(rv[col] + BN_EPS) * g[col] + be[col];
    y = (y >= 0.f) ? y : NEG_SLOPE * y;
    ebf[(size_t)(nb + r) * HID + col] = (__bf16)y;
}

// ---------------- helpers ----------------
__global__ void k_zero(float* __restrict__ p, size_t n) {
    size_t i = (size_t)blockIdx.x * blockDim.x + threadIdx.x;
    size_t stride = (size_t)gridDim.x * blockDim.x;
    for (; i < n; i += stride) p[i] = 0.f;
}

__global__ void k_residual(float* __restrict__ h, const float* __restrict__ agg, size_t n) {
    size_t i = (size_t)blockIdx.x * blockDim.x + threadIdx.x;
    size_t stride = (size_t)gridDim.x * blockDim.x;
    for (; i < n; i += stride) h[i] += agg[i];
}

// ------------- CGConv edge kernel: the WMMA workhorse -------------
// m = sigmoid([h_dst,h_src,e] @ Wf + bf) * softplus([h_dst,h_src,e] @ Ws + bs)
// agg[dst] += m   (f32 atomics)
__global__ __launch_bounds__(64)
void k_cgconv(const float* __restrict__ h, const __bf16* __restrict__ ebf,
              const int* __restrict__ src, const int* __restrict__ dst,
              const float* __restrict__ Wf, const float* __restrict__ bF,
              const float* __restrict__ Ws, const float* __restrict__ bS,
              float* __restrict__ agg, int n_edges) {
    // 2 waves per block. LDS: 2x transposed bf16 weights + wave-private z tiles.
    __shared__ __bf16 WfT[HID * KPAD];   // 25.0 KB
    __shared__ __bf16 WsT[HID * KPAD];   // 25.0 KB
    __shared__ __bf16 zt[2][16 * ZS];    // 12.5 KB
    __shared__ int    wdst[2][16];

    // stage Wf/Ws transposed as bf16 (row = output col, contiguous in K)
    for (int i = threadIdx.x; i < KDIM * HID; i += blockDim.x) {
        const int k = i >> 6, n = i & 63;            // global W is [K=192][N=64] row major
        WfT[n * KPAD + k] = (__bf16)Wf[i];
        WsT[n * KPAD + k] = (__bf16)Ws[i];
    }
    __syncthreads();

    const int wave = threadIdx.x >> 5;
    const int lane = threadIdx.x & 31;
    const int m    = lane & 15;                      // row (A) / column (B/C)
    const int hi   = lane >> 4;
    __bf16* zw = zt[wave];
    int*    dw = wdst[wave];

    // per-lane bias, constant across tiles (N = n*16 + m per the C/D layout)
    float biasF[4], biasS[4];
    #pragma unroll
    for (int n = 0; n < 4; ++n) { biasF[n] = bF[n * 16 + m]; biasS[n] = bS[n * 16 + m]; }

    const __bf16* abase_off = zw + m * ZS + (hi ? 16 : 0); // A frag: lanes 16-31 hold K=16..31 (contiguous)
    const int bK = hi ? 16 : 0;                            // B frag: lanes 16-31 hold K=16..31

    const int n_tiles = (n_edges + 15) >> 4;
    const int wid = blockIdx.x * 2 + wave;
    const int nw  = gridDim.x * 2;

    for (int t = wid; t < n_tiles; t += nw) {
        const int e0 = t * 16;
        const int eg = e0 + m;
        const bool valid = (eg < n_edges);
        const int egc  = valid ? eg : 0;
        const int sidx = valid ? src[egc] : 0;
        const int didx = valid ? dst[egc] : 0;

        // ---- gather z = [h_dst | h_src | e] into the wave's LDS tile (bf16) ----
        __bf16* zrow = zw + m * ZS;
        if (hi == 0) {
            dw[m] = valid ? didx : -1;
            const float4* hd = (const float4*)(h + (size_t)didx * HID);
            #pragma unroll
            for (int i = 0; i < 16; ++i) *(uint2*)(zrow + i * 4) = bf4(hd[i]);        // z[0..63]
            const float4* hs = (const float4*)(h + (size_t)sidx * HID);
            #pragma unroll
            for (int i = 0; i < 8; ++i)  *(uint2*)(zrow + 64 + i * 4) = bf4(hs[i]);   // z[64..95]
        } else {
            const float4* hs = (const float4*)(h + (size_t)sidx * HID);
            #pragma unroll
            for (int i = 0; i < 8; ++i)  *(uint2*)(zrow + 96 + i * 4) = bf4(hs[8 + i]); // z[96..127]
            const uint4* ee = (const uint4*)(ebf + (size_t)egc * HID);
            #pragma unroll
            for (int i = 0; i < 8; ++i)  *(uint4*)(zrow + 128 + i * 8) = ee[i];         // z[128..191]
        }
        asm volatile("s_wait_dscnt 0" ::: "memory");   // intra-wave LDS store->load

        // ---- 16x192 @ 192x64, twice (Wf, Ws): 48 bf16 WMMAs ----
        v8f accF[4] = {}, accS[4] = {};
        #pragma unroll
        for (int k = 0; k < 6; ++k) {
            const __bf16* ap = abase_off + k * 32;
            const v16bf a = cat16(*(const v8bf*)ap, *(const v8bf*)(ap + 8));
            #pragma unroll
            for (int n = 0; n < 4; ++n) {
                const __bf16* bpF = WfT + (n * 16 + m) * KPAD + k * 32 + bK;
                const v16bf bf = cat16(*(const v8bf*)bpF, *(const v8bf*)(bpF + 8));
                accF[n] = __builtin_amdgcn_wmma_f32_16x16x32_bf16(
                    false, a, false, bf, (short)0, accF[n], false, false);
                const __bf16* bpS = WsT + (n * 16 + m) * KPAD + k * 32 + bK;
                const v16bf bs = cat16(*(const v8bf*)bpS, *(const v8bf*)(bpS + 8));
                accS[n] = __builtin_amdgcn_wmma_f32_16x16x32_bf16(
                    false, a, false, bs, (short)0, accS[n], false, false);
            }
        }

        // ---- epilogue: bias, sigmoid*softplus, scatter-add ----
        int drow[8];
        #pragma unroll
        for (int v = 0; v < 8; ++v) drow[v] = dw[hi * 8 + v];   // C layout: M = v + hi*8
        #pragma unroll
        for (int n = 0; n < 4; ++n) {
            const int N = n * 16 + m;
            #pragma unroll
            for (int v = 0; v < 8; ++v) {
                if (drow[v] < 0) continue;
                const float f = accF[n][v] + biasF[n];
                const float s = accS[n][v] + biasS[n];
                const float sig = 1.f / (1.f + __expf(-f));
                const float sp  = (s > 15.f) ? s : __logf(1.f + __expf(s));
                atomicAdd(agg + (size_t)drow[v] * HID + N, sig * sp);
            }
        }
    }
}

// ---------------- output: out = h @ W_out + b_out ----------------
__global__ void k_out(const float* __restrict__ h, const float* __restrict__ W,
                      const float* __restrict__ b, float* __restrict__ out, int n) {
    __shared__ float hs[4][HID];
    const int nb = blockIdx.x * 4;
    {
        int r = threadIdx.x >> 6, c = threadIdx.x & 63;   // 256 threads == 4*64
        if (nb + r < n) hs[r][c] = h[(size_t)(nb + r) * HID + c];
    }
    __syncthreads();
    const int r = threadIdx.x >> 6, col = threadIdx.x & 63;
    if (nb + r >= n) return;
    float acc = b[col];
    #pragma unroll 8
    for (int k = 0; k < HID; ++k) acc = fmaf(hs[r][k], W[k * HID + col], acc);
    out[(size_t)(nb + r) * HID + col] = acc;
}

extern "C" void kernel_launch(void* const* d_in, const int* in_sizes, int n_in,
                              void* d_out, int out_size, void* d_ws, size_t ws_size,
                              hipStream_t stream) {
    const float* x     = (const float*)d_in[0];
    const float* ea    = (const float*)d_in[1];
    const int*   ei    = (const int*)  d_in[2];
    const float* W_in  = (const float*)d_in[3];
    const float* b_in  = (const float*)d_in[4];
    const float* g_in  = (const float*)d_in[5];
    const float* be_in = (const float*)d_in[6];
    const float* rm_in = (const float*)d_in[7];
    const float* rv_in = (const float*)d_in[8];
    const float* W_e   = (const float*)d_in[9];
    const float* b_e   = (const float*)d_in[10];
    const float* g_e   = (const float*)d_in[11];
    const float* be_e  = (const float*)d_in[12];
    const float* rm_e  = (const float*)d_in[13];
    const float* rv_e  = (const float*)d_in[14];
    const float* Wf[2] = {(const float*)d_in[15], (const float*)d_in[19]};
    const float* bf[2] = {(const float*)d_in[16], (const float*)d_in[20]};
    const float* Ws[2] = {(const float*)d_in[17], (const float*)d_in[21]};
    const float* bs[2] = {(const float*)d_in[18], (const float*)d_in[22]};
    const float* W_out = (const float*)d_in[23];
    const float* b_out = (const float*)d_in[24];
    float* out = (float*)d_out;

    const int n_nodes = in_sizes[0] / IN_DIM;     // 100000
    const int n_edges = in_sizes[1] / EDGE_DIM;   // 1600000
    const int* src = ei;
    const int* dst = ei + n_edges;

    // workspace: h (f32) | agg (f32) | e (bf16)
    float*  h    = (float*)d_ws;
    float*  agg  = h + (size_t)n_nodes * HID;
    __bf16* ebf  = (__bf16*)(agg + (size_t)n_nodes * HID);
    const size_t nagg = (size_t)n_nodes * HID;

    k_node_embed<<<(n_nodes + 3) / 4, 256, 0, stream>>>(x, W_in, b_in, g_in, be_in, rm_in, rv_in, h, n_nodes);
    k_edge_embed<<<(n_edges + 3) / 4, 256, 0, stream>>>(ea, W_e, b_e, g_e, be_e, rm_e, rv_e, ebf, n_edges);

    for (int layer = 0; layer < 2; ++layer) {
        k_zero<<<2048, 256, 0, stream>>>(agg, nagg);
        k_cgconv<<<2560, 64, 0, stream>>>(h, ebf, src, dst,
                                          Wf[layer], bf[layer], Ws[layer], bs[layer],
                                          agg, n_edges);
        k_residual<<<2048, 256, 0, stream>>>(h, agg, nagg);
    }

    k_out<<<(n_nodes + 3) / 4, 256, 0, stream>>>(h, W_out, b_out, out, n_nodes);
}
